// BlockwiseKronLinear_45921790329285
// MI455X (gfx1250) — compile-verified, matrix operand order checked by reading, not run on hardware
//
#include <hip/hip_runtime.h>
#include <stdint.h>

// Problem constants (from reference)
#define RANK   16
#define A1d    64
#define A2d    8
#define B1d    64
#define B2d    8
#define BATCH  32768
#define KDIM   (A1d * B1d)      // 4096
#define NDIM   (A2d * B2d)      // 64
#define KCHUNK 32               // K per WMMA
#define KCHUNKS (KDIM / KCHUNK) // 128
#define NTILES  (NDIM / 16)     // 4

typedef __bf16 bf16_t;
typedef bf16_t          v16bf __attribute__((ext_vector_type(16)));
typedef float           v16f  __attribute__((ext_vector_type(16)));
typedef float           v8f   __attribute__((ext_vector_type(8)));
typedef float           v4f   __attribute__((ext_vector_type(4)));
typedef unsigned short  v16u  __attribute__((ext_vector_type(16)));

union V16 { v16bf bf; v16u us; };

// scalar RNE f32 -> bf16 bits (used only in the tiny pack kernel)
static __device__ __forceinline__ unsigned short f2bf_rne(float f) {
    unsigned u = __float_as_uint(f);
    u += 0x7FFFu + ((u >> 16) & 1u);
    return (unsigned short)(u >> 16);
}
static __device__ __forceinline__ float bf2f(unsigned short h) {
    return __uint_as_float(((unsigned)h) << 16);
}

// ---------------------------------------------------------------------------
// Kernel 1: build w = sum_r kron(s*a[r], b[r]), split into bf16 hi/lo, and
// store directly in V_WMMA_F32_16X16X32_BF16 B-fragment order:
//   fragment index = kchunk*NTILES + ntile, 512 bf16 per fragment,
//   element = lane*16 + p  with
//     n      = ntile*16 + (lane & 15)
//     klocal = ((p < 8) ? p : p + 8) + (lane >> 4)*8   // VGPR0-3: K 0..15, 4-7: K 16..31
// ---------------------------------------------------------------------------
__global__ void __launch_bounds__(256)
pack_w_kernel(const float* __restrict__ s,
              const float* __restrict__ a,
              const float* __restrict__ b,
              unsigned short* __restrict__ whi,
              unsigned short* __restrict__ wlo) {
    const int e = blockIdx.x * blockDim.x + threadIdx.x;   // 0 .. KDIM*NDIM-1
    const int frag   = e >> 9;
    const int lane   = (e >> 4) & 31;
    const int p      = e & 15;
    const int kchunk = frag >> 2;
    const int nt     = frag & 3;

    const int klocal = ((p < 8) ? p : p + 8) + ((lane >> 4) << 3);
    const int k = kchunk * KCHUNK + klocal;                // w row  (i*B1 + kk)
    const int n = nt * 16 + (lane & 15);                   // w col  (j*B2 + l)

    const int i  = k >> 6, kk = k & 63;
    const int j  = n >> 3, l  = n & 7;

    const float sv = s[i * A2d + j];
    float acc = 0.f;
#pragma unroll
    for (int r = 0; r < RANK; ++r) {
        const float av = a[(r * A1d + i) * A2d + j];
        const float bv = b[(r * B1d + kk) * B2d + l];
        acc = fmaf(sv * av, bv, acc);
    }
    const unsigned short hi = f2bf_rne(acc);
    whi[e] = hi;
    wlo[e] = f2bf_rne(acc - bf2f(hi));
}

// ---------------------------------------------------------------------------
// Kernel 2: out = x @ w + bias using bf16x3-split V_WMMA_F32_16X16X32_BF16.
// One wave (wave32) per 16-row M tile; 4 N tiles (full N=64) per wave.
// x is loaded non-temporally (pure stream, don't evict B fragments from WGP$);
// B fragments use default RT hints so all 8 waves of a block hit WGP$/L2.
// ---------------------------------------------------------------------------
__global__ void __launch_bounds__(256)
gemm_kernel(const float* __restrict__ x,
            const unsigned short* __restrict__ whi,
            const unsigned short* __restrict__ wlo,
            const float* __restrict__ bias,
            float* __restrict__ out) {
    const int lane  = threadIdx.x & 31;
    const int wave  = threadIdx.x >> 5;
    const int mtile = blockIdx.x * 8 + wave;               // 0 .. 2047
    const int m0    = mtile * 16;
    const int half  = lane >> 4;                           // K-half select
    const int row   = m0 + (lane & 15);                    // A-matrix row for this lane
    const float* __restrict__ xrow = x + (size_t)row * KDIM;

    v8f acc[NTILES] = {};                                  // 4 x 8 VGPR f32 accumulators

    for (int kc = 0; kc < KCHUNKS; ++kc) {
        const int kbase = kc * KCHUNK + half * 8;

        // Prefetch x two K-chunks ahead (global_prefetch_b8)
        if (kc + 2 < KCHUNKS)
            __builtin_prefetch(xrow + kbase + 2 * KCHUNK, 0, 1);

        // A fragment: lane needs k = kbase..kbase+7 (VGPR0-3) and
        //             k = kbase+16..kbase+23 (VGPR4-7); two contiguous 32B runs.
        // Non-temporal: x is a 512MB stream, keep it out of WGP$.
        const v4f x0 = __builtin_nontemporal_load((const v4f*)(xrow + kbase));
        const v4f x1 = __builtin_nontemporal_load((const v4f*)(xrow + kbase + 4));
        const v4f x2 = __builtin_nontemporal_load((const v4f*)(xrow + kbase + 16));
        const v4f x3 = __builtin_nontemporal_load((const v4f*)(xrow + kbase + 20));
        v16f xv;
#pragma unroll
        for (int t = 0; t < 4; ++t) {
            xv[t]      = x0[t];
            xv[4 + t]  = x1[t];
            xv[8 + t]  = x2[t];
            xv[12 + t] = x3[t];
        }

        // bf16x3 split with native packed converts (RNE):
        //   xh = bf16(x);  xl = bf16(x - f32(xh))
        V16 ahi, alo;
        ahi.bf = __builtin_convertvector(xv, v16bf);
        const v16f hif = __builtin_convertvector(ahi.bf, v16f);
        const v16f lof = xv - hif;
        alo.bf = __builtin_convertvector(lof, v16bf);

#pragma unroll
        for (int nt = 0; nt < NTILES; ++nt) {
            const size_t off = ((size_t)(kc * NTILES + nt) << 9) + (size_t)lane * 16;
            V16 bhi, blo;
            bhi.us = *(const v16u*)(whi + off);            // 32B contiguous per lane (WGP$/L2 hit)
            blo.us = *(const v16u*)(wlo + off);

            // x@w = xh@wh + xh@wl + xl@wh   (drop tiny xl@wl term)
            acc[nt] = __builtin_amdgcn_wmma_f32_16x16x32_bf16(
                false, ahi.bf, false, bhi.bf, (short)0, acc[nt], false, false);
            acc[nt] = __builtin_amdgcn_wmma_f32_16x16x32_bf16(
                false, ahi.bf, false, blo.bf, (short)0, acc[nt], false, false);
            acc[nt] = __builtin_amdgcn_wmma_f32_16x16x32_bf16(
                false, alo.bf, false, bhi.bf, (short)0, acc[nt], false, false);
        }
    }

    // Epilogue: C layout = VGPR r -> M = r (lanes 0-15) / r+8 (lanes 16-31),
    //           lane%16 -> N.  Add bias, store f32.
#pragma unroll
    for (int nt = 0; nt < NTILES; ++nt) {
        const int col = nt * 16 + (lane & 15);
        const float bv = bias[col];
#pragma unroll
        for (int r = 0; r < 8; ++r) {
            const int orow = m0 + half * 8 + r;
            out[(size_t)orow * NDIM + col] = acc[nt][r] + bv;
        }
    }
}

// ---------------------------------------------------------------------------
extern "C" void kernel_launch(void* const* d_in, const int* in_sizes, int n_in,
                              void* d_out, int out_size, void* d_ws, size_t ws_size,
                              hipStream_t stream) {
    (void)in_sizes; (void)n_in; (void)out_size; (void)ws_size;
    const float* x    = (const float*)d_in[0];
    const float* s    = (const float*)d_in[1];
    const float* a    = (const float*)d_in[2];
    const float* b    = (const float*)d_in[3];
    const float* bias = (const float*)d_in[4];

    unsigned short* whi = (unsigned short*)d_ws;                  // 512 KB
    unsigned short* wlo = whi + (size_t)KDIM * NDIM;              // 512 KB

    // Build w in WMMA B-fragment layout (tiny: 262144 elements)
    pack_w_kernel<<<(KDIM * NDIM) / 256, 256, 0, stream>>>(s, a, b, whi, wlo);

    // Main GEMM: 2048 M-tiles, 8 waves/block -> 256 blocks
    gemm_kernel<<<BATCH / 128, 256, 0, stream>>>(x, whi, wlo, bias, (float*)d_out);
}
